// CLIPEncoderLayer_79345225826619
// MI455X (gfx1250) — compile-verified, matrix operand order checked by reading, not run on hardware
//
#include <hip/hip_runtime.h>

// ---------------- constants ----------------
#define Dm   1024
#define Hm   16
#define HDm  64
#define FFm  4096
#define Bm   8
#define Lm   1024
#define Mrows (Bm * Lm)        // 8192
#define EPSc 1e-5f

typedef __attribute__((ext_vector_type(16))) __bf16 v16bf;
typedef __attribute__((ext_vector_type(8)))  float  v8f;
typedef __attribute__((ext_vector_type(4)))  unsigned int u32x4;
typedef __attribute__((ext_vector_type(8)))  int          i32x8;
typedef __attribute__((ext_vector_type(4)))  int          i32x4;

#if defined(__has_builtin)
#if __has_builtin(__builtin_amdgcn_tensor_load_to_lds) && \
    __has_builtin(__builtin_amdgcn_s_wait_tensorcnt)
#define USE_TDM 1
#endif
#endif
#ifndef USE_TDM
#define USE_TDM 0
#endif

union V16 {
    uint4 u4[2];
    unsigned short s[16];
    v16bf v;
};

__device__ __forceinline__ unsigned short f2bf(float f) {
    unsigned int u = __float_as_uint(f);
    unsigned int r = u + 0x7fffu + ((u >> 16) & 1u);   // round to nearest even
    return (unsigned short)(r >> 16);
}

#if USE_TDM
// Issue one TDM 2D tile load: tile_rows x 32 bf16 elements, row stride K
// elements, into LDS with +16B padding per 64B row (=> row pitch 80B).
__device__ __forceinline__ void tdm_tile_load(const unsigned short* gsrc,
                                              unsigned int lds_off,
                                              int K, int rows_total,
                                              int tile_rows) {
    unsigned long long ga = (unsigned long long)(size_t)gsrc;
    u32x4 g0;
    g0[0] = 1u;                                   // count=1, user mode
    g0[1] = lds_off;                              // lds_addr (bytes)
    g0[2] = (unsigned int)ga;                     // global_addr[31:0]
    g0[3] = ((unsigned int)(ga >> 32) & 0x01FFFFFFu) | 0x80000000u; // type=2

    i32x8 g1;
    // data_size=1 (2B) | pad_enable | pad_interval=3 (16 DW) | pad_amount=3 (4 DW)
    g1[0] = (int)(0x00010000u | (1u << 20) | (3u << 22) | (3u << 25));
    g1[1] = (int)(((unsigned)K & 0xFFFFu) << 16);          // tensor_dim0[15:0]
    g1[2] = (int)((((unsigned)K >> 16) & 0xFFFFu) |
                  (((unsigned)rows_total & 0xFFFFu) << 16)); // dim0 hi | dim1 lo
    g1[3] = (int)((((unsigned)rows_total >> 16) & 0xFFFFu) |
                  (32u << 16));                            // dim1 hi | tile_dim0=32
    g1[4] = (int)((unsigned)tile_rows & 0xFFFFu);          // tile_dim1 | tile_dim2=0
    g1[5] = (int)(unsigned)K;                              // tensor_dim0_stride lo
    g1[6] = 0;                                             // stride hi | dim1_stride lo
    g1[7] = 0;

    i32x4 z4 = {0, 0, 0, 0};
#if __clang_major__ >= 23
    i32x8 z8 = {0, 0, 0, 0, 0, 0, 0, 0};
    __builtin_amdgcn_tensor_load_to_lds(g0, g1, z4, z4, z8, 0);
#else
    __builtin_amdgcn_tensor_load_to_lds(g0, g1, z4, z4, 0);
#endif
}
#endif

// ---------------- fp32 -> bf16 transpose-convert: dst[N][K] = src[K][N] ----------------
__global__ __launch_bounds__(256)
void cvt_t_kernel(const float* __restrict__ src, unsigned short* __restrict__ dst,
                  int K, int N) {
    __shared__ unsigned short tile[32][33];
    int n0 = blockIdx.x * 32, k0 = blockIdx.y * 32;
    int tx = threadIdx.x, ty = threadIdx.y;   // block (32,8)
    #pragma unroll
    for (int i = 0; i < 4; i++) {
        int r = ty + i * 8;
        tile[r][tx] = f2bf(src[(size_t)(k0 + r) * N + n0 + tx]);
    }
    __syncthreads();
    #pragma unroll
    for (int i = 0; i < 4; i++) {
        int r = ty + i * 8;
        dst[(size_t)(n0 + r) * K + k0 + tx] = tile[tx][r];
    }
}

// ---------------- LayerNorm -> bf16 ----------------
__global__ __launch_bounds__(256)
void ln_bf16_kernel(const float* __restrict__ x,
                    const float* __restrict__ gamma,
                    const float* __restrict__ beta,
                    unsigned short* __restrict__ out) {
    int row = blockIdx.x;
    int t = threadIdx.x;
    __shared__ float red[256];
    const float* xr = x + (size_t)row * Dm;

    float sum = 0.f;
    for (int i = t; i < Dm; i += 256) sum += xr[i];
    red[t] = sum; __syncthreads();
    for (int st = 128; st > 0; st >>= 1) {
        if (t < st) red[t] += red[t + st];
        __syncthreads();
    }
    float mu = red[0] * (1.0f / Dm);
    __syncthreads();

    float vs = 0.f;
    for (int i = t; i < Dm; i += 256) { float d = xr[i] - mu; vs += d * d; }
    red[t] = vs; __syncthreads();
    for (int st = 128; st > 0; st >>= 1) {
        if (t < st) red[t] += red[t + st];
        __syncthreads();
    }
    float inv = rsqrtf(red[0] * (1.0f / Dm) + EPSc);
    __syncthreads();

    unsigned short* orow = out + (size_t)row * Dm;
    for (int i = t; i < Dm; i += 256)
        orow[i] = f2bf((xr[i] - mu) * inv * gamma[i] + beta[i]);
}

// ---------------- bf16 WMMA GEMM, B pre-transposed, double-buffered TDM ----------------
// C[M,N] = A[M,K] * Bt[N,K]^T  (+bias) (act) (+resid) (*oscale)
// Block tile 64(M) x 128(N), BK=32, 4 waves, each wave 32x64.
// heads: 0 = plain, 1 = [B,H,L,HD], 2 = [B,H,HD,L] (transposed per head)
__global__ __launch_bounds__(128)
void gemm_bf16_kernel(const unsigned short* __restrict__ A,
                      const unsigned short* __restrict__ Bt,
                      const float* __restrict__ bias,
                      const float* __restrict__ resid,
                      float* __restrict__ outF,
                      unsigned short* __restrict__ outB,
                      int M, int N, int K,
                      int act, int heads, float oscale) {
    __shared__ __align__(16) unsigned short As[2][64][40];   // [m][k], 80B pitch
    __shared__ __align__(16) unsigned short Bs[2][128][40];  // [n][k], 80B pitch

    int t    = threadIdx.x;
    int w    = t >> 5, lane = t & 31;
    int wr   = w >> 1, wc = w & 1;
    int mloc = lane & 15, half = lane >> 4;
    int m0 = blockIdx.y * 64, n0 = blockIdx.x * 128;

    v8f c[2][4] = {};
    int nsteps = K >> 5;

#if USE_TDM
    unsigned int ldsA[2], ldsB[2];
    ldsA[0] = (unsigned int)(size_t)(void*)&As[0][0][0];
    ldsA[1] = (unsigned int)(size_t)(void*)&As[1][0][0];
    ldsB[0] = (unsigned int)(size_t)(void*)&Bs[0][0][0];
    ldsB[1] = (unsigned int)(size_t)(void*)&Bs[1][0][0];

    if (w == 0) {   // prologue: stage 0 DMAs (wave-uniform; TDM ignores EXEC)
        tdm_tile_load(&A[(size_t)m0 * K],  ldsA[0], K, M, 64);
        tdm_tile_load(&Bt[(size_t)n0 * K], ldsB[0], K, N, 128);
    }

    for (int s = 0; s < nsteps; s++) {
        int p = s & 1;
        if (w == 0) {
            if (s + 1 < nsteps) {   // prefetch next stage into other buffer
                int k1 = (s + 1) << 5;
                tdm_tile_load(&A[(size_t)m0 * K + k1],  ldsA[p ^ 1], K, M, 64);
                tdm_tile_load(&Bt[(size_t)n0 * K + k1], ldsB[p ^ 1], K, N, 128);
                __builtin_amdgcn_s_wait_tensorcnt(2);  // stage-s pair landed
            } else {
                __builtin_amdgcn_s_wait_tensorcnt(0);
            }
        }
        __syncthreads();

        V16 af[2], bf[4];
        #pragma unroll
        for (int mt = 0; mt < 2; mt++) {
            int row = wr * 32 + mt * 16 + mloc;
            af[mt].u4[0] = *(const uint4*)&As[p][row][half * 8];
            af[mt].u4[1] = *(const uint4*)&As[p][row][16 + half * 8];
        }
        #pragma unroll
        for (int nt = 0; nt < 4; nt++) {
            int col = wc * 64 + nt * 16 + mloc;
            bf[nt].u4[0] = *(const uint4*)&Bs[p][col][half * 8];
            bf[nt].u4[1] = *(const uint4*)&Bs[p][col][16 + half * 8];
        }
        #pragma unroll
        for (int mt = 0; mt < 2; mt++)
            #pragma unroll
            for (int nt = 0; nt < 4; nt++)
                c[mt][nt] = __builtin_amdgcn_wmma_f32_16x16x32_bf16(
                    false, af[mt].v, false, bf[nt].v, (short)0, c[mt][nt],
                    false, false);
        __syncthreads();
    }
#else
    for (int s = 0; s < nsteps; s++) {
        int k0 = s << 5;
        int seg = t & 3;
        int rr  = t >> 2;
        #pragma unroll
        for (int r = 0; r < 2; r++) {
            int row = rr + r * 32;
            uint4 av = *(const uint4*)&A[(size_t)(m0 + row) * K + k0 + seg * 8];
            *(uint4*)&As[0][row][seg * 8] = av;
        }
        #pragma unroll
        for (int r = 0; r < 4; r++) {
            int row = rr + r * 32;
            uint4 bv = *(const uint4*)&Bt[(size_t)(n0 + row) * K + k0 + seg * 8];
            *(uint4*)&Bs[0][row][seg * 8] = bv;
        }
        __syncthreads();

        V16 af[2], bf[4];
        #pragma unroll
        for (int mt = 0; mt < 2; mt++) {
            int row = wr * 32 + mt * 16 + mloc;
            af[mt].u4[0] = *(const uint4*)&As[0][row][half * 8];
            af[mt].u4[1] = *(const uint4*)&As[0][row][16 + half * 8];
        }
        #pragma unroll
        for (int nt = 0; nt < 4; nt++) {
            int col = wc * 64 + nt * 16 + mloc;
            bf[nt].u4[0] = *(const uint4*)&Bs[0][col][half * 8];
            bf[nt].u4[1] = *(const uint4*)&Bs[0][col][16 + half * 8];
        }
        #pragma unroll
        for (int mt = 0; mt < 2; mt++)
            #pragma unroll
            for (int nt = 0; nt < 4; nt++)
                c[mt][nt] = __builtin_amdgcn_wmma_f32_16x16x32_bf16(
                    false, af[mt].v, false, bf[nt].v, (short)0, c[mt][nt],
                    false, false);
        __syncthreads();
    }
#endif

    // ---- epilogue ----
    #pragma unroll
    for (int mt = 0; mt < 2; mt++) {
        #pragma unroll
        for (int nt = 0; nt < 4; nt++) {
            int col = n0 + wc * 64 + nt * 16 + mloc;
            #pragma unroll
            for (int i = 0; i < 8; i++) {
                int row = m0 + wr * 32 + mt * 16 + i + 8 * half;
                float v = c[mt][nt][i];
                if (bias)  v += bias[col];
                if (act == 1)
                    v = v * __builtin_amdgcn_rcpf(1.0f + __expf(-1.702f * v));
                if (resid) v += resid[(size_t)row * N + col];
                v *= oscale;
                if (outF) outF[(size_t)row * N + col] = v;
                if (outB) {
                    size_t idx;
                    int bb = row >> 10, ll = row & 1023;
                    int hh = col >> 6,  dd = col & 63;
                    if (heads == 1)
                        idx = (((size_t)(bb * Hm + hh) * Lm) + ll) * HDm + dd;
                    else if (heads == 2)
                        idx = (((size_t)(bb * Hm + hh) * HDm) + dd) * Lm + ll;
                    else
                        idx = (size_t)row * N + col;
                    outB[idx] = f2bf(v);
                }
            }
        }
    }
}

// ---------------- flash attention (one wave / 16 query rows / head) ----------------
// q,k: bf16 [B*H][L][HD] (q pre-scaled by HD^-0.5); v: bf16 [B*H][HD][L].
// out: bf16 [B*L][D].
__global__ __launch_bounds__(32)
void attn_kernel(const unsigned short* __restrict__ q,
                 const unsigned short* __restrict__ k,
                 const unsigned short* __restrict__ v,
                 const int* __restrict__ mask,
                 unsigned short* __restrict__ out) {
    int bh = blockIdx.x;              // 0..127
    int b  = bh >> 4, h = bh & 15;
    int q0 = blockIdx.y * 16;
    int lane = threadIdx.x;
    int mloc = lane & 15, half = lane >> 4;

    const unsigned short* Q  = q + (size_t)bh * Lm * HDm;
    const unsigned short* Kp = k + (size_t)bh * Lm * HDm;
    const unsigned short* Vt = v + (size_t)bh * HDm * Lm;

    // Q A-fragments for d = 0..31 and 32..63
    V16 aq[2];
    #pragma unroll
    for (int s = 0; s < 2; s++) {
        int d0 = s * 32;
        aq[s].u4[0] = *(const uint4*)&Q[(size_t)(q0 + mloc) * HDm + d0 + half * 8];
        aq[s].u4[1] = *(const uint4*)&Q[(size_t)(q0 + mloc) * HDm + d0 + 16 + half * 8];
    }

    float run_max[8], run_sum[8];
    #pragma unroll
    for (int i = 0; i < 8; i++) { run_max[i] = -1e30f; run_sum[i] = 0.f; }
    v8f o[4] = {};

    __shared__ __align__(16) unsigned short pl[16][40]; // P tile 16x32

    for (int kt = 0; kt < Lm / 32; kt++) {
        int kb = kt * 32;
        // ---- scores: two 16-key subtiles, K-depth = 64 (2 wmma each) ----
        v8f sc[2] = {};
        #pragma unroll
        for (int sub = 0; sub < 2; sub++) {
            int key = kb + sub * 16 + mloc;
            #pragma unroll
            for (int s = 0; s < 2; s++) {
                int d0 = s * 32;
                V16 bk;
                bk.u4[0] = *(const uint4*)&Kp[(size_t)key * HDm + d0 + half * 8];
                bk.u4[1] = *(const uint4*)&Kp[(size_t)key * HDm + d0 + 16 + half * 8];
                sc[sub] = __builtin_amdgcn_wmma_f32_16x16x32_bf16(
                    false, aq[s].v, false, bk.v, (short)0, sc[sub], false, false);
            }
        }
        // mask (per key column)
        float mk0 = (mask[b * Lm + kb + mloc]      > 0) ? 0.f : -1e30f;
        float mk1 = (mask[b * Lm + kb + 16 + mloc] > 0) ? 0.f : -1e30f;
        #pragma unroll
        for (int i = 0; i < 8; i++) { sc[0][i] += mk0; sc[1][i] += mk1; }

        // ---- online softmax ----
        float p0[8], p1[8];
        #pragma unroll
        for (int i = 0; i < 8; i++) {
            float tm = fmaxf(sc[0][i], sc[1][i]);
            #pragma unroll
            for (int off = 1; off < 16; off <<= 1)
                tm = fmaxf(tm, __shfl_xor(tm, off, 32));
            float nm = fmaxf(run_max[i], tm);
            float corr = __expf(run_max[i] - nm);
            run_max[i] = nm;
            p0[i] = __expf(sc[0][i] - nm);
            p1[i] = __expf(sc[1][i] - nm);
            float ts = p0[i] + p1[i];
            #pragma unroll
            for (int off = 1; off < 16; off <<= 1)
                ts += __shfl_xor(ts, off, 32);
            run_sum[i] = run_sum[i] * corr + ts;
            #pragma unroll
            for (int f = 0; f < 4; f++) o[f][i] *= corr;
        }

        // ---- P -> LDS (re-layout D-frag -> A-frag) ----
        #pragma unroll
        for (int i = 0; i < 8; i++) {
            pl[i + 8 * half][mloc]      = f2bf(p0[i]);
            pl[i + 8 * half][16 + mloc] = f2bf(p1[i]);
        }
        __syncthreads();
        V16 ap;
        ap.u4[0] = *(const uint4*)&pl[mloc][half * 8];
        ap.u4[1] = *(const uint4*)&pl[mloc][16 + half * 8];
        __syncthreads();

        // ---- O += P * V  (4 wmma; V fragments are contiguous rows of Vt) ----
        #pragma unroll
        for (int f = 0; f < 4; f++) {
            V16 bv;
            int d = f * 16 + mloc;
            bv.u4[0] = *(const uint4*)&Vt[(size_t)d * Lm + kb + half * 8];
            bv.u4[1] = *(const uint4*)&Vt[(size_t)d * Lm + kb + 16 + half * 8];
            o[f] = __builtin_amdgcn_wmma_f32_16x16x32_bf16(
                false, ap.v, false, bv.v, (short)0, o[f], false, false);
        }
    }

    // ---- normalize + store: out[b*L + q][h*64 + d] ----
    #pragma unroll
    for (int f = 0; f < 4; f++) {
        #pragma unroll
        for (int i = 0; i < 8; i++) {
            int qq  = q0 + i + 8 * half;
            int col = h * HDm + f * 16 + mloc;
            out[(size_t)(b * Lm + qq) * Dm + col] = f2bf(o[f][i] / run_sum[i]);
        }
    }
}

// ---------------- host-side launch ----------------
extern "C" void kernel_launch(void* const* d_in, const int* in_sizes, int n_in,
                              void* d_out, int out_size, void* d_ws, size_t ws_size,
                              hipStream_t stream) {
    const float* x     = (const float*)d_in[0];
    const int*   amask = (const int*)  d_in[1];
    const float* wq = (const float*)d_in[2];  const float* bq = (const float*)d_in[3];
    const float* wk = (const float*)d_in[4];  const float* bk = (const float*)d_in[5];
    const float* wv = (const float*)d_in[6];  const float* bv = (const float*)d_in[7];
    const float* wo = (const float*)d_in[8];  const float* bo = (const float*)d_in[9];
    const float* ln1s = (const float*)d_in[10]; const float* ln1b = (const float*)d_in[11];
    const float* ln2s = (const float*)d_in[12]; const float* ln2b = (const float*)d_in[13];
    const float* w1 = (const float*)d_in[14]; const float* b1 = (const float*)d_in[15];
    const float* w2 = (const float*)d_in[16]; const float* b2 = (const float*)d_in[17];
    float* outp = (float*)d_out;

    char* base = (char*)d_ws;
    size_t off = 0;
    auto alloc = [&](size_t bytes) -> char* {
        char* p = base + off;
        off = (off + bytes + 255) & ~(size_t)255;
        return p;
    };
    // transposed bf16 weights: [N][K]
    unsigned short* wqb = (unsigned short*)alloc((size_t)Dm * Dm * 2);
    unsigned short* wkb = (unsigned short*)alloc((size_t)Dm * Dm * 2);
    unsigned short* wvb = (unsigned short*)alloc((size_t)Dm * Dm * 2);
    unsigned short* wob = (unsigned short*)alloc((size_t)Dm * Dm * 2);
    unsigned short* w1b = (unsigned short*)alloc((size_t)Dm * FFm * 2);
    unsigned short* w2b = (unsigned short*)alloc((size_t)FFm * Dm * 2);
    unsigned short* h1b = (unsigned short*)alloc((size_t)Mrows * Dm * 2);
    unsigned short* qb  = (unsigned short*)alloc((size_t)Mrows * Dm * 2);
    unsigned short* kb  = (unsigned short*)alloc((size_t)Mrows * Dm * 2);
    unsigned short* vb  = (unsigned short*)alloc((size_t)Mrows * Dm * 2);
    unsigned short* ab  = (unsigned short*)alloc((size_t)Mrows * Dm * 2);
    float*          x1  = (float*)        alloc((size_t)Mrows * Dm * 4);
    unsigned short* h2b = (unsigned short*)alloc((size_t)Mrows * Dm * 2);
    unsigned short* m1b = (unsigned short*)alloc((size_t)Mrows * FFm * 2);
    (void)ws_size; (void)n_in; (void)in_sizes; (void)out_size;

    const float qscale = 0.125f; // HD^-0.5
    dim3 tb(32, 8);

    // weights -> transposed bf16  (src is [K][N])
    cvt_t_kernel<<<dim3(Dm / 32, Dm / 32),  tb, 0, stream>>>(wq, wqb, Dm, Dm);
    cvt_t_kernel<<<dim3(Dm / 32, Dm / 32),  tb, 0, stream>>>(wk, wkb, Dm, Dm);
    cvt_t_kernel<<<dim3(Dm / 32, Dm / 32),  tb, 0, stream>>>(wv, wvb, Dm, Dm);
    cvt_t_kernel<<<dim3(Dm / 32, Dm / 32),  tb, 0, stream>>>(wo, wob, Dm, Dm);
    cvt_t_kernel<<<dim3(FFm / 32, Dm / 32), tb, 0, stream>>>(w1, w1b, Dm, FFm);
    cvt_t_kernel<<<dim3(Dm / 32, FFm / 32), tb, 0, stream>>>(w2, w2b, FFm, Dm);

    // LN1
    ln_bf16_kernel<<<Mrows, 256, 0, stream>>>(x, ln1s, ln1b, h1b);

    // QKV projections (Q pre-scaled; V transposed per head)
    dim3 gDD(Dm / 128, Mrows / 64);
    gemm_bf16_kernel<<<gDD, 128, 0, stream>>>(h1b, wqb, bq, nullptr, nullptr, qb,
                                              Mrows, Dm, Dm, 0, 1, qscale);
    gemm_bf16_kernel<<<gDD, 128, 0, stream>>>(h1b, wkb, bk, nullptr, nullptr, kb,
                                              Mrows, Dm, Dm, 0, 1, 1.0f);
    gemm_bf16_kernel<<<gDD, 128, 0, stream>>>(h1b, wvb, bv, nullptr, nullptr, vb,
                                              Mrows, Dm, Dm, 0, 2, 1.0f);

    // attention
    attn_kernel<<<dim3(Bm * Hm, Lm / 16), 32, 0, stream>>>(qb, kb, vb, amask, ab);

    // output projection + residual -> x1
    gemm_bf16_kernel<<<gDD, 128, 0, stream>>>(ab, wob, bo, x, x1, nullptr,
                                              Mrows, Dm, Dm, 0, 0, 1.0f);

    // LN2
    ln_bf16_kernel<<<Mrows, 256, 0, stream>>>(x1, ln2s, ln2b, h2b);

    // MLP up + quickGELU -> bf16
    dim3 gDF(FFm / 128, Mrows / 64);
    gemm_bf16_kernel<<<gDF, 128, 0, stream>>>(h2b, w1b, b1, nullptr, nullptr, m1b,
                                              Mrows, FFm, Dm, 1, 0, 1.0f);

    // MLP down + residual -> d_out
    gemm_bf16_kernel<<<gDD, 128, 0, stream>>>(m1b, w2b, b2, x1, outp, nullptr,
                                              Mrows, Dm, FFm, 0, 0, 1.0f);
}
